// RPN_85332410237305
// MI455X (gfx1250) — matrix operand order repaired and checked by reading, not compiled, not run on hardware
//
#include <hip/hip_runtime.h>

// ---------------- types for WMMA ----------------
typedef __attribute__((ext_vector_type(16))) __bf16 v16bf;
typedef __attribute__((ext_vector_type(8)))  float  v8f;

struct Half32 { uint4 lo, hi; };   // 32 bytes == v16bf

// ---------------- constants ----------------
#define BATCH   8
#define FH      50
#define FW      50
#define CIN     512
#define COUT    512
#define NANCH   9
#define NPOS    (FH*FW)            // 2500
#define NBOX    (NPOS*NANCH)       // 22500
#define PRE_NMS 6000
#define POST_NMS 300
#define NMS_TH  0.7f
#define MIN_SZ  16.0f
#define IMG     800.0f

// d_out regions (float/int32 elements)
#define REG_OFF   0                      // 8*22500*4 = 720000
#define CLS_OFF   720000                 // 8*22500*2 = 360000
#define ROIS_OFF  1080000                // 8*300*4   = 9600
#define ROIID_OFF 1089600                // 8*300     = 2400
#define ANCH_OFF  1092000                // 22500*4   = 90000

// workspace byte offsets (all 64B aligned); total need ~49.4 MB
#define WS_XB   ((size_t)0)                       // 10.24M bf16 = 20,480,000 B
#define WS_WB   ((size_t)20480000)                // 9*512*512 bf16 = 4,718,592 B
#define WS_SH   ((size_t)25198592)                // 10.24M bf16 = 20,480,000 B
#define WS_BOX  ((size_t)45678592)                // 8*22500*4 f32 = 2,880,000 B
#define WS_SCO  ((size_t)48558592)                // 8*22500 f32 = 720,000 B
#define WS_WH   ((size_t)49278592)                // 64*512 bf16 = 65,536 B

__device__ __forceinline__ unsigned short f2bf(float f) {
  unsigned u = __builtin_bit_cast(unsigned, f);
  return (unsigned short)((u + 0x7FFFu + ((u >> 16) & 1u)) >> 16);   // RNE
}
__device__ __forceinline__ float bf2f(unsigned short h) {
  unsigned u = ((unsigned)h) << 16;
  return __builtin_bit_cast(float, u);
}
__device__ __forceinline__ unsigned fkey(float f) {  // order-preserving key
  unsigned u = __builtin_bit_cast(unsigned, f);
  return (u & 0x80000000u) ? ~u : (u | 0x80000000u);
}

// ---------- K0: x NCHW f32 -> NHWC bf16 ----------
__global__ void k_cvt_x(const float* __restrict__ x, unsigned short* __restrict__ xb) {
  int i = blockIdx.x * blockDim.x + threadIdx.x;
  if (i >= BATCH * CIN * NPOS) return;
  int b = i / (CIN * NPOS);
  int r = i % (CIN * NPOS);
  int c = r / NPOS;
  int p = r % NPOS;
  xb[((size_t)b * NPOS + p) * CIN + c] = f2bf(x[i]);
}

// ---------- K1: W_share OIHW f32 -> bf16 in B-fragment order ----------
// layout: block = chunk*32 + ntile  (chunk = tap*16 + c0/32), then [lane][i]
// lane l holds N = ntile*16 + l%16, K(i) = i + (l>=16 ? 16 : 0)
__global__ void k_cvt_w(const float* __restrict__ w, unsigned short* __restrict__ wb) {
  int o = blockIdx.x * blockDim.x + threadIdx.x;
  if (o >= 9 * COUT * CIN) return;
  int block = o >> 9, r = o & 511;
  int lane = r >> 4, i = r & 15;
  int ntile = block & 31, chunk = block >> 5;
  int tap = chunk >> 4, c0 = (chunk & 15) << 5;
  int n = ntile * 16 + (lane & 15);
  int c = c0 + i + ((lane >= 16) ? 16 : 0);
  wb[o] = f2bf(w[((size_t)n * CIN + c) * 9 + tap]);
}

// ---------- K1b: head weights (cls 18 + reg 36, zero-pad to 64) -> bf16 B-frags ----------
__global__ void k_cvt_whead(const float* __restrict__ Wc, const float* __restrict__ Wr,
                            unsigned short* __restrict__ wh) {
  int o = blockIdx.x * blockDim.x + threadIdx.x;
  if (o >= 64 * CIN) return;
  int block = o >> 9, r = o & 511;
  int lane = r >> 4, i = r & 15;
  int ntile = block & 3, chunk = block >> 2;
  int c = (chunk << 5) + i + ((lane >= 16) ? 16 : 0);
  int n = ntile * 16 + (lane & 15);
  float v = 0.f;
  if (n < 18)      v = Wc[(size_t)n * CIN + c];
  else if (n < 54) v = Wr[(size_t)(n - 18) * CIN + c];
  wh[o] = f2bf(v);
}

// ---------- K2: implicit-GEMM 3x3 conv via WMMA bf16, bias+ReLU ----------
// 2x2 register tile per wave: (2 M-tiles = 32 x-positions) x (2 N-tiles = 32 ch).
// grid.x = 8*50*2 (32-wide x strips), grid.y = 2 (8 waves x 2 ntiles x 16 ch)
__global__ __launch_bounds__(256) void k_conv_wmma(
    const unsigned short* __restrict__ xb, const unsigned short* __restrict__ wb,
    const float* __restrict__ bias, unsigned short* __restrict__ shd) {
  int bx = blockIdx.x;
  int b = bx / (FH * 2);
  int rem = bx % (FH * 2);
  int y = rem >> 1;
  int x0a = (rem & 1) << 5;         // first M-tile origin
  int x0b = x0a + 16;               // second M-tile origin
  int wave = threadIdx.x >> 5;
  int lane = threadIdx.x & 31;
  int nt0 = (blockIdx.y * 8 + wave) * 2;   // this wave owns ntiles nt0, nt0+1
  int ncol = lane & 15;             // column N (and A-row m) per ISA §7.12.2
  int hi = lane >> 4;               // K-half select

  v8f acc00 = {0.f, 0.f, 0.f, 0.f, 0.f, 0.f, 0.f, 0.f};
  v8f acc01 = acc00, acc10 = acc00, acc11 = acc00;

  for (int tap = 0; tap < 9; ++tap) {
    int dy = tap / 3 - 1, dx = tap % 3 - 1;
    int yy = y + dy;
    bool yok = (yy >= 0) & (yy < FH);
    int xsa = x0a + ncol, xca = xsa + dx;
    int xsb = x0b + ncol, xcb = xsb + dx;
    bool ava = yok & (xca >= 0) & (xca < FW) & (xsa < FW);
    bool avb = yok & (xcb >= 0) & (xcb < FW) & (xsb < FW);
    const unsigned short* apa =
        xb + (((size_t)(b * FH + (ava ? yy : 0)) * FW + (ava ? xca : 0)) * CIN + hi * 16);
    const unsigned short* apb =
        xb + (((size_t)(b * FH + (avb ? yy : 0)) * FW + (avb ? xcb : 0)) * CIN + hi * 16);
#pragma unroll
    for (int c0 = 0; c0 < CIN; c0 += 32) {
      Half32 sa, sc;
      if (ava) { sa.lo = *(const uint4*)(apa + c0); sa.hi = *(const uint4*)(apa + c0 + 8); }
      else     { sa.lo = make_uint4(0,0,0,0); sa.hi = make_uint4(0,0,0,0); }
      if (avb) { sc.lo = *(const uint4*)(apb + c0); sc.hi = *(const uint4*)(apb + c0 + 8); }
      else     { sc.lo = make_uint4(0,0,0,0); sc.hi = make_uint4(0,0,0,0); }
      v16bf aA = __builtin_bit_cast(v16bf, sa);
      v16bf aB = __builtin_bit_cast(v16bf, sc);
      int chunk = tap * 16 + (c0 >> 5);
      const unsigned short* wbase = wb + (((size_t)chunk * 32) * 512 + lane * 16);
      const uint4* bp0 = (const uint4*)(wbase + (size_t)nt0 * 512);
      const uint4* bp1 = (const uint4*)(wbase + (size_t)(nt0 + 1) * 512);
      // prefetch next chunk's weights (gfx1250 global_prefetch path)
      __builtin_prefetch(wbase + (size_t)16384 + (size_t)nt0 * 512, 0, 1);
      Half32 sb0; sb0.lo = bp0[0]; sb0.hi = bp0[1];   // 32 contiguous bytes/lane
      Half32 sb1; sb1.lo = bp1[0]; sb1.hi = bp1[1];
      v16bf bf0 = __builtin_bit_cast(v16bf, sb0);
      v16bf bf1 = __builtin_bit_cast(v16bf, sb1);
      acc00 = __builtin_amdgcn_wmma_f32_16x16x32_bf16(false, aA, false, bf0, (short)0, acc00, false, false);
      acc01 = __builtin_amdgcn_wmma_f32_16x16x32_bf16(false, aA, false, bf1, (short)0, acc01, false, false);
      acc10 = __builtin_amdgcn_wmma_f32_16x16x32_bf16(false, aB, false, bf0, (short)0, acc10, false, false);
      acc11 = __builtin_amdgcn_wmma_f32_16x16x32_bf16(false, aB, false, bf1, (short)0, acc11, false, false);
    }
  }

  int n0 = nt0 * 16 + ncol;
  int n1 = n0 + 16;
  float bn0 = bias[n0];
  float bn1 = bias[n1];
  size_t rowbase = (size_t)(b * FH + y) * FW;
#pragma unroll
  for (int r = 0; r < 8; ++r) {
    int mm = r + hi * 8;                 // D row M = r + 8*(lane>=16)
    int xoa = x0a + mm;                  // first M-tile: always < 50
    {
      size_t o = (rowbase + xoa) * COUT;
      float v0 = acc00[r] + bn0; v0 = v0 > 0.f ? v0 : 0.f;
      float v1 = acc01[r] + bn1; v1 = v1 > 0.f ? v1 : 0.f;
      shd[o + n0] = f2bf(v0);
      shd[o + n1] = f2bf(v1);
    }
    int xob = x0b + mm;
    if (xob < FW) {
      size_t o = (rowbase + xob) * COUT;
      float v0 = acc10[r] + bn0; v0 = v0 > 0.f ? v0 : 0.f;
      float v1 = acc11[r] + bn1; v1 = v1 > 0.f ? v1 : 0.f;
      shd[o + n0] = f2bf(v0);
      shd[o + n1] = f2bf(v1);
    }
  }
}

// ---------- K3: 1x1 heads via WMMA (M=16 positions, N=64 ch, K=512) + softmax ----------
__global__ __launch_bounds__(128) void k_heads_wmma(
    const unsigned short* __restrict__ shd, const unsigned short* __restrict__ wh,
    const float* __restrict__ bc, const float* __restrict__ br,
    float* __restrict__ outReg, float* __restrict__ outCls,
    float* __restrict__ scores) {
  int tile = blockIdx.x;                  // 16 consecutive flat positions
  int wave = threadIdx.x >> 5;            // ntile 0..3
  int lane = threadIdx.x & 31;
  int ncol = lane & 15;
  int hi = lane >> 4;

  v8f acc = {0.f, 0.f, 0.f, 0.f, 0.f, 0.f, 0.f, 0.f};
  const unsigned short* ap =
      shd + (((size_t)tile * 16 + ncol) * CIN + hi * 16);
#pragma unroll
  for (int c0 = 0; c0 < CIN; c0 += 32) {
    Half32 sa;
    sa.lo = *(const uint4*)(ap + c0);
    sa.hi = *(const uint4*)(ap + c0 + 8);
    v16bf afrag = __builtin_bit_cast(v16bf, sa);
    int chunk = c0 >> 5;
    const uint4* bp =
        (const uint4*)(wh + (((size_t)(chunk * 4 + wave)) * 512 + lane * 16));
    Half32 sb; sb.lo = bp[0]; sb.hi = bp[1];
    v16bf bfrag = __builtin_bit_cast(v16bf, sb);
    acc = __builtin_amdgcn_wmma_f32_16x16x32_bf16(
        false, afrag, false, bfrag, (short)0, acc, false, false);
  }

  int n = wave * 16 + ncol;               // output channel 0..63
  float bn = 0.f;
  if (n < 18)      bn = bc[n];
  else if (n < 54) bn = br[n - 18];

#pragma unroll
  for (int r = 0; r < 8; ++r) {
    int m = r + hi * 8;
    int pos = tile * 16 + m;
    int b = pos / NPOS, p = pos % NPOS;
    float v = acc[r] + bn;
    float o = __shfl_xor(v, 1, 32);       // partner channel n^1 (same row)
    if (n < 18) {                          // cls logits -> pairwise softmax
      float mx = fmaxf(v, o);
      float e  = __expf(v - mx);
      float eo = __expf(o - mx);
      float prob = e / (e + eo);
      int a = n >> 1, j = n & 1;
      size_t ci = (size_t)b * NBOX + (size_t)p * NANCH + a;
      outCls[ci * 2 + j] = prob;
      if (j == 1) scores[ci] = prob;      // fg score
    } else if (n < 54) {                   // reg output
      int oc = n - 18;
      int a = oc >> 2, j = oc & 3;
      outReg[((size_t)b * NBOX + (size_t)p * NANCH + a) * 4 + j] = v;
    }
  }
}

// ---------- K4: anchors + box decode + clip + min-size mask ----------
__global__ void k_decode(const float* __restrict__ reg, float* __restrict__ scores,
                         float* __restrict__ boxes, float* __restrict__ anch) {
  int i = blockIdx.x * blockDim.x + threadIdx.x;
  if (i >= BATCH * NBOX) return;
  int b = i / NBOX, idx = i % NBOX;
  int p = idx / NANCH, a = idx % NANCH;
  int y = p / FW, x = p % FW;
  int ri = a / 3, si = a % 3;
  float rr = (ri == 0) ? 0.5f : ((ri == 1) ? 1.0f : 2.0f);
  float ss = (si == 0) ? 8.0f : ((si == 1) ? 16.0f : 32.0f);
  float ha = 16.0f * ss * sqrtf(rr);
  float wa = 16.0f * ss * sqrtf(1.0f / rr);
  float cya = y * 16.0f + 8.0f, cxa = x * 16.0f + 8.0f;
  if (b == 0) {
    anch[idx * 4 + 0] = cya - 0.5f * ha;
    anch[idx * 4 + 1] = cxa - 0.5f * wa;
    anch[idx * 4 + 2] = cya + 0.5f * ha;
    anch[idx * 4 + 3] = cxa + 0.5f * wa;
  }
  const float* r4 = reg + (size_t)i * 4;
  float cy = r4[0] * ha + cya, cx = r4[1] * wa + cxa;
  float hh = __expf(r4[2]) * ha, ww = __expf(r4[3]) * wa;
  float y1 = fminf(fmaxf(cy - 0.5f * hh, 0.f), IMG);
  float x1 = fminf(fmaxf(cx - 0.5f * ww, 0.f), IMG);
  float y2 = fminf(fmaxf(cy + 0.5f * hh, 0.f), IMG);
  float x2 = fminf(fmaxf(cx + 0.5f * ww, 0.f), IMG);
  boxes[(size_t)i * 4 + 0] = y1;
  boxes[(size_t)i * 4 + 1] = x1;
  boxes[(size_t)i * 4 + 2] = y2;
  boxes[(size_t)i * 4 + 3] = x2;
  bool keep = ((y2 - y1) >= MIN_SZ) & ((x2 - x1) >= MIN_SZ);
  float fg = scores[i];
  scores[i] = keep ? fg : -__builtin_inff();
}

// ---------- K5: per-batch top-6000 threshold (bisection on sortable keys) ----------
__global__ __launch_bounds__(1024) void k_topk(float* __restrict__ scores) {
  int b = blockIdx.x, t = threadIdx.x;
  float* s = scores + (size_t)b * NBOX;
  __shared__ int red[1024];
  unsigned lo = 0u, hi = 0xFFFFFFFFu;
  for (int it = 0; it < 33 && lo < hi; ++it) {
    unsigned d = hi - lo;
    unsigned mid = lo + (d >> 1) + (d & 1u);     // ceil midpoint
    int c = 0;
    for (int i = t; i < NBOX; i += 1024) c += (fkey(s[i]) >= mid) ? 1 : 0;
    red[t] = c; __syncthreads();
    for (int off = 512; off > 0; off >>= 1) {
      if (t < off) red[t] += red[t + off];
      __syncthreads();
    }
    int total = red[0]; __syncthreads();
    if (total >= PRE_NMS) lo = mid; else hi = mid - 1;
  }
  unsigned V = lo;                               // key of the 6000th score
  for (int i = t; i < NBOX; i += 1024)
    if (fkey(s[i]) < V) s[i] = -__builtin_inff();
}

// ---------- K6: per-batch sequential NMS scan (300 picks) ----------
__global__ __launch_bounds__(1024) void k_nms(
    const float* __restrict__ boxes_all, float* __restrict__ scores,
    float* __restrict__ rois, int* __restrict__ roiid) {
  int b = blockIdx.x, t = threadIdx.x;
  const float* bx = boxes_all + (size_t)b * NBOX * 4;
  float* s = scores + (size_t)b * NBOX;
  __shared__ float sv[1024];
  __shared__ int   si_[1024];
  __shared__ float sbox[4];
  for (int it = 0; it < POST_NMS; ++it) {
    float bv = -__builtin_inff(); int bi = 0;
    for (int i = t; i < NBOX; i += 1024) {
      float v = s[i];
      if (v > bv) { bv = v; bi = i; }
    }
    sv[t] = bv; si_[t] = bi; __syncthreads();
    for (int off = 512; off > 0; off >>= 1) {
      if (t < off) {
        float v2 = sv[t + off]; int i2 = si_[t + off];
        if (v2 > sv[t] || (v2 == sv[t] && i2 < si_[t])) { sv[t] = v2; si_[t] = i2; }
      }
      __syncthreads();
    }
    if (t == 0) {
      int ii = si_[0];
      int valid = (sv[0] > -__builtin_inff()) ? 1 : 0;
      float b0 = bx[ii * 4 + 0], b1 = bx[ii * 4 + 1];
      float b2 = bx[ii * 4 + 2], b3 = bx[ii * 4 + 3];
      sbox[0] = b0; sbox[1] = b1; sbox[2] = b2; sbox[3] = b3;
      float fv = valid ? 1.0f : 0.0f;
      size_t ro = ((size_t)b * POST_NMS + it) * 4;
      rois[ro + 0] = b0 * fv; rois[ro + 1] = b1 * fv;
      rois[ro + 2] = b2 * fv; rois[ro + 3] = b3 * fv;
      roiid[b * POST_NMS + it] = valid ? b : 0;
    }
    __syncthreads();
    float a0 = sbox[0], a1 = sbox[1], a2 = sbox[2], a3 = sbox[3];
    float area1 = (a2 - a0) * (a3 - a1);
    for (int i = t; i < NBOX; i += 1024) {
      float c0 = bx[i * 4 + 0], c1 = bx[i * 4 + 1];
      float c2 = bx[i * 4 + 2], c3 = bx[i * 4 + 3];
      float ty1 = fmaxf(a0, c0), tx1 = fmaxf(a1, c1);
      float ty2 = fminf(a2, c2), tx2 = fminf(a3, c3);
      float inter = fmaxf(ty2 - ty1, 0.f) * fmaxf(tx2 - tx1, 0.f);
      float area2 = (c2 - c0) * (c3 - c1);
      float iou = inter / (area1 + area2 - inter + 1e-9f);
      if (iou > NMS_TH) s[i] = -__builtin_inff();
    }
    __syncthreads();
  }
}

extern "C" void kernel_launch(void* const* d_in, const int* in_sizes, int n_in,
                              void* d_out, int out_size, void* d_ws, size_t ws_size,
                              hipStream_t stream) {
  const float* x   = (const float*)d_in[0];
  const float* Ws  = (const float*)d_in[1];
  const float* bs  = (const float*)d_in[2];
  const float* Wc  = (const float*)d_in[3];
  const float* bc  = (const float*)d_in[4];
  const float* Wr  = (const float*)d_in[5];
  const float* br  = (const float*)d_in[6];
  float* out = (float*)d_out;
  char* ws = (char*)d_ws;

  unsigned short* xb  = (unsigned short*)(ws + WS_XB);
  unsigned short* wb  = (unsigned short*)(ws + WS_WB);
  unsigned short* shd = (unsigned short*)(ws + WS_SH);
  float* boxes  = (float*)(ws + WS_BOX);
  float* scores = (float*)(ws + WS_SCO);
  unsigned short* wh  = (unsigned short*)(ws + WS_WH);

  float* outReg  = out + REG_OFF;
  float* outCls  = out + CLS_OFF;
  float* outRois = out + ROIS_OFF;
  int*   outRoiId = (int*)(out + ROIID_OFF);
  float* outAnch = out + ANCH_OFF;

  // K0: x -> bf16 NHWC
  {
    int n = BATCH * CIN * NPOS;
    k_cvt_x<<<(n + 255) / 256, 256, 0, stream>>>(x, xb);
  }
  // K1: W_share -> bf16, B-fragment swizzled
  {
    int n = 9 * COUT * CIN;
    k_cvt_w<<<(n + 255) / 256, 256, 0, stream>>>(Ws, wb);
  }
  // K1b: head weights -> bf16 B-fragments (64x512)
  {
    int n = 64 * CIN;
    k_cvt_whead<<<(n + 255) / 256, 256, 0, stream>>>(Wc, Wr, wh);
  }
  // K2: WMMA conv + bias + ReLU (2x2 register tile per wave)
  k_conv_wmma<<<dim3(BATCH * FH * 2, 2), 256, 0, stream>>>(xb, wb, bs, shd);
  // K3: WMMA 1x1 heads + pairwise softmax via lane-pair shuffle
  k_heads_wmma<<<(BATCH * NPOS) / 16, 128, 0, stream>>>(shd, wh, bc, br,
                                                        outReg, outCls, scores);
  // K4: anchors + decode
  {
    int n = BATCH * NBOX;
    k_decode<<<(n + 255) / 256, 256, 0, stream>>>(outReg, scores, boxes, outAnch);
  }
  // K5: top-6000 threshold per batch
  k_topk<<<BATCH, 1024, 0, stream>>>(scores);
  // K6: NMS per batch
  k_nms<<<BATCH, 1024, 0, stream>>>(boxes, scores, outRois, outRoiId);
}
